// SCAM_54674933678855
// MI455X (gfx1250) — matrix operand order repaired and checked by reading, not compiled, not run on hardware
//
#include <hip/hip_runtime.h>
#include <hip/hip_bf16.h>

// ---------------------------------------------------------------------------
// SCAM cross-attention, MI455X (gfx1250, wave32, WMMA f32<=f16x f16).
// Shapes: B=2, h=128, w=256, c=64. 256 slices of 256x256 attention.
//
// d_out layout (floats): cross[8388608] | cost0[16777216] | cost1[16777216]
// d_ws  layout: Qh (f16, 16MB) | Vt (f16, 16MB) | stats (f32, 1MB)  ~= 33MB
// ---------------------------------------------------------------------------

typedef _Float16 h16 __attribute__((ext_vector_type(16)));
typedef _Float16 h8  __attribute__((ext_vector_type(8)));
typedef _Float16 h4  __attribute__((ext_vector_type(4)));
typedef float    f8  __attribute__((ext_vector_type(8)));
typedef float    f4  __attribute__((ext_vector_type(4)));

union HV { h16 v; h8 h[2]; };

static __device__ __forceinline__ f8 wmma16(h16 a, h16 b, f8 c) {
  // D = A(16x32 f16) x B(32x16 f16) + C(16x16 f32)
  return __builtin_amdgcn_wmma_f32_16x16x32_f16(
      /*neg_a=*/false, a, /*neg_b=*/false, b,
      /*c_mod=*/(short)0, c, /*reuse_a=*/false, /*reuse_b=*/false);
}

static __device__ __forceinline__ f8 zero8() {
  f8 z;
#pragma unroll
  for (int i = 0; i < 8; ++i) z[i] = 0.0f;
  return z;
}

// A-matrix loader: 16x32 f16 tile, row-major source (row = lane&15).
// Lanes 0-15 hold K {0..15}; lanes 16-31 hold K {16..31}.
// `row` points at element (m, 0) of the K-block; kbA = (lane<16)?0:8.
static __device__ __forceinline__ h16 load_a16(const _Float16* row, int kbA) {
  HV u;
  u.h[0] = *(const h8*)(row + 2 * kbA);
  u.h[1] = *(const h8*)(row + 2 * kbA + 8);
  return u.v;
}

#define C0_OFF    8388608
#define C1_OFF    25165824
#define QH_HALVES 8388608   // 2 sides * 256 slices * 256*64

// ---------------------------------------------------------------------------
// K1: LayerNorm + pointwise projections.
//   Qh[side][s][j][c]  = (LN(x) W1^T + b1)  (f16, row-major)
//   Vt[side][s][c][j]  = (x W2^T + b2)      (f16, TRANSPOSED c-major)
// grid 512 (side*256 + slice), block 256 (thread = pixel j in slice)
// ---------------------------------------------------------------------------
__global__ __launch_bounds__(256) void k1_proj(
    const float* __restrict__ x,
    const float* __restrict__ nlw, const float* __restrict__ nlb,
    const float* __restrict__ nrw, const float* __restrict__ nrb,
    const float* __restrict__ wl1, const float* __restrict__ bl1,
    const float* __restrict__ wr1, const float* __restrict__ br1,
    const float* __restrict__ wl2, const float* __restrict__ bl2,
    const float* __restrict__ wr2, const float* __restrict__ br2,
    _Float16* __restrict__ qh, _Float16* __restrict__ vt)
{
  __shared__ float W1[4096], W2[4096];
  __shared__ float LW[64], LB[64], B1[64], B2[64];

  const int side = blockIdx.x >> 8;
  const int s    = blockIdx.x & 255;
  const int j    = threadIdx.x;

  const float* w1 = side ? wr1 : wl1;
  const float* w2 = side ? wr2 : wl2;
  for (int k = j; k < 4096; k += 256) { W1[k] = w1[k]; W2[k] = w2[k]; }
  if (j < 64) {
    LW[j] = side ? nrw[j] : nlw[j];
    LB[j] = side ? nrb[j] : nlb[j];
    B1[j] = side ? br1[j] : bl1[j];
    B2[j] = side ? br2[j] : bl2[j];
  }
  __syncthreads();

  const float* xr = x + ((size_t)side * 65536 + (size_t)s * 256 + j) * 64;
  float xv[64];
#pragma unroll
  for (int c = 0; c < 64; ++c) xv[c] = xr[c];

  // V = x W2^T + b2, stored transposed (c-major) for WMMA B-operand loads.
  _Float16* vcol = vt + (size_t)side * (QH_HALVES / 2) + (size_t)s * 16384 + j;
#pragma unroll 2
  for (int o = 0; o < 64; ++o) {
    float a = B2[o];
    const float* wrow = &W2[o * 64];
#pragma unroll
    for (int c = 0; c < 64; ++c) a = fmaf(xv[c], wrow[c], a);
    vcol[o * 256] = (_Float16)a;
  }

  // LayerNorm in place
  float mu = 0.0f;
#pragma unroll
  for (int c = 0; c < 64; ++c) mu += xv[c];
  mu *= (1.0f / 64.0f);
  float var = 0.0f;
#pragma unroll
  for (int c = 0; c < 64; ++c) { float d = xv[c] - mu; var = fmaf(d, d, var); }
  var *= (1.0f / 64.0f);
  const float is = rsqrtf(var + 1e-6f);
#pragma unroll
  for (int c = 0; c < 64; ++c) xv[c] = (xv[c] - mu) * is * LW[c] + LB[c];

  // Q = LN(x) W1^T + b1, row-major f16, written 16B at a time.
  h8* qd = (h8*)(qh + (size_t)side * (QH_HALVES / 2) + (size_t)s * 16384 +
                 (size_t)j * 64);
#pragma unroll 2
  for (int p = 0; p < 8; ++p) {
    h8 t;
#pragma unroll
    for (int q = 0; q < 8; ++q) {
      const int o = p * 8 + q;
      float a = B1[o];
      const float* wrow = &W1[o * 64];
#pragma unroll
      for (int c = 0; c < 64; ++c) a = fmaf(xv[c], wrow[c], a);
      t[q] = (_Float16)a;
    }
    qd[p] = t;
  }
}

// ---------------------------------------------------------------------------
// K2: scores. One workgroup per slice (256 blocks, 8 waves).
//   A = scale * Ql Qr^T  (WMMA);  cost0 = cost0_in + A; cost1 = cost1_in + A^T
//   stats: max0/sum0 = row max/sumexp of A; max1/sum1 = col max/sumexp.
//   A is recomputed (WMMA) for the stats pass -- FLOP-free by roofline.
// Column partials accumulate into per-wave LDS rows (single writer per slot,
// deterministic); row partials reduce via shuffles (rows owned by one wave).
// ---------------------------------------------------------------------------
__global__ __launch_bounds__(256) void k2_scores(
    const _Float16* __restrict__ qh,
    const float* __restrict__ c0in, const float* __restrict__ c1in,
    float* __restrict__ out, float* __restrict__ stats)
{
  __shared__ float cmW[8 * 256];   // per-wave column partials (max, then sum)
  __shared__ float scol[256];      // combined column max

  const int s    = blockIdx.x;
  const int tid  = threadIdx.x;
  const int lane = tid & 31;
  const int wv   = tid >> 5;
  const int r15  = lane & 15;
  const int hi   = lane >> 4;
  const int kbA  = hi ? 8 : 0;
  const int kbB  = hi ? 16 : 0;
  const float scale = 0.125f;  // c^-0.5

  const _Float16* Ql = qh + (size_t)s * 16384;
  const _Float16* Qr = qh + (QH_HALVES / 2) + (size_t)s * 16384;
  float* c0o = out + C0_OFF + (size_t)s * 65536;
  float* c1o = out + C1_OFF + (size_t)s * 65536;
  const float* c0i = c0in + (size_t)s * 65536;
  const float* c1i = c1in + (size_t)s * 65536;
  float* gmax0 = stats + (size_t)s * 256;
  float* gsum0 = stats + 65536 + (size_t)s * 256;
  float* gmax1 = stats + 131072 + (size_t)s * 256;
  float* gsum1 = stats + 196608 + (size_t)s * 256;

  for (int k = tid; k < 2048; k += 256) cmW[k] = -3.0e38f;
  __syncthreads();

  float rmax[2][8];

  // ---- phase 1: compute A, write cost0/cost1, row/col maxima -------------
  for (int tt = 0; tt < 2; ++tt) {
    const int ti = wv * 2 + tt;
    const _Float16* arow = Ql + (size_t)(ti * 16 + r15) * 64;
    const h16 a0 = load_a16(arow, kbA);
    const h16 a1 = load_a16(arow + 32, kbA);
    float rm[8];
#pragma unroll
    for (int r = 0; r < 8; ++r) rm[r] = -3.0e38f;

    for (int tj = 0; tj < 16; ++tj) {
      const _Float16* brow = Qr + (size_t)(tj * 16 + r15) * 64;
      const h16 b0 = *(const h16*)(brow + kbB);
      const h16 b1 = *(const h16*)(brow + 32 + kbB);
      f8 acc = zero8();
      acc = wmma16(a0, b0, acc);
      acc = wmma16(a1, b1, acc);
      const int n = tj * 16 + r15;
      float v[8];
#pragma unroll
      for (int r = 0; r < 8; ++r) v[r] = acc[r] * scale;

      // cost0: strided per-row scalar stores (coalesced across lanes)
#pragma unroll
      for (int r = 0; r < 8; ++r) {
        const int m = ti * 16 + hi * 8 + r;
        c0o[m * 256 + n] = v[r] + c0i[m * 256 + n];
      }
      // cost1: 8 consecutive floats per lane -> 2x b128
      {
        const int cb = n * 256 + ti * 16 + hi * 8;
        const f4 i0 = *(const f4*)(c1i + cb);
        const f4 i1 = *(const f4*)(c1i + cb + 4);
        f4 o0, o1;
#pragma unroll
        for (int r = 0; r < 4; ++r) { o0[r] = v[r] + i0[r]; o1[r] = v[r + 4] + i1[r]; }
        *(f4*)(c1o + cb) = o0;
        *(f4*)(c1o + cb + 4) = o1;
      }
      // maxima
      float cm = v[0];
#pragma unroll
      for (int r = 0; r < 8; ++r) { rm[r] = fmaxf(rm[r], v[r]); }
#pragma unroll
      for (int r = 1; r < 8; ++r) cm = fmaxf(cm, v[r]);
      cm = fmaxf(cm, __shfl_xor(cm, 16));
      if (hi == 0) {  // single writer per column slot -> deterministic
        float* p = &cmW[wv * 256 + n];
        *p = fmaxf(*p, cm);
      }
    }
    // deterministic row-max reduction across the 16 lanes of each half
#pragma unroll
    for (int r = 0; r < 8; ++r) {
      float v = rm[r];
      v = fmaxf(v, __shfl_xor(v, 1));
      v = fmaxf(v, __shfl_xor(v, 2));
      v = fmaxf(v, __shfl_xor(v, 4));
      v = fmaxf(v, __shfl_xor(v, 8));
      rmax[tt][r] = v;
    }
    if (r15 == 0) {
#pragma unroll
      for (int r = 0; r < 8; ++r) gmax0[ti * 16 + hi * 8 + r] = rmax[tt][r];
    }
  }
  __syncthreads();
  // combine column maxima over the 8 waves
  {
    float m = cmW[tid];
#pragma unroll
    for (int w2 = 1; w2 < 8; ++w2) m = fmaxf(m, cmW[w2 * 256 + tid]);
    scol[tid] = m;
    gmax1[tid] = m;
  }
  __syncthreads();
  for (int k = tid; k < 2048; k += 256) cmW[k] = 0.0f;  // reuse for sums
  __syncthreads();

  // ---- phase 2: recompute A (WMMA), accumulate exp sums ------------------
  for (int tt = 0; tt < 2; ++tt) {
    const int ti = wv * 2 + tt;
    const _Float16* arow = Ql + (size_t)(ti * 16 + r15) * 64;
    const h16 a0 = load_a16(arow, kbA);
    const h16 a1 = load_a16(arow + 32, kbA);
    float rs[8];
#pragma unroll
    for (int r = 0; r < 8; ++r) rs[r] = 0.0f;

    for (int tj = 0; tj < 16; ++tj) {
      const _Float16* brow = Qr + (size_t)(tj * 16 + r15) * 64;
      const h16 b0 = *(const h16*)(brow + kbB);
      const h16 b1 = *(const h16*)(brow + 32 + kbB);
      f8 acc = zero8();
      acc = wmma16(a0, b0, acc);
      acc = wmma16(a1, b1, acc);
      const int n = tj * 16 + r15;
      const float cmv = scol[n];
      float cs = 0.0f;
#pragma unroll
      for (int r = 0; r < 8; ++r) {
        const float v = acc[r] * scale;
        rs[r] += __expf(v - rmax[tt][r]);
        cs += __expf(v - cmv);
      }
      cs += __shfl_xor(cs, 16);
      if (hi == 0) cmW[wv * 256 + n] += cs;  // single writer, deterministic
    }
#pragma unroll
    for (int r = 0; r < 8; ++r) {
      float v = rs[r];
      v += __shfl_xor(v, 1);
      v += __shfl_xor(v, 2);
      v += __shfl_xor(v, 4);
      v += __shfl_xor(v, 8);
      rs[r] = v;
    }
    if (r15 == 0) {
#pragma unroll
      for (int r = 0; r < 8; ++r) gsum0[ti * 16 + hi * 8 + r] = rs[r];
    }
  }
  __syncthreads();
  {
    float sum = 0.0f;
#pragma unroll
    for (int w2 = 0; w2 < 8; ++w2) sum += cmW[w2 * 256 + tid];
    gsum1[tid] = sum;
  }
}

// ---------------------------------------------------------------------------
// K3: softmax-weighted PV + fused epilogue. grid 512 = inst*256 + slice.
//   inst 0: out_l = x_l + rowsoftmax(cost0) @ V_r * beta
//   inst 1: out_r = x_r + rowsoftmax(cost1) @ V_l * gamma   (== axis=-2 path)
// P tiles normalized to f16 in LDS per 64-wide K-chunk, WMMA-accumulated.
// ---------------------------------------------------------------------------
__global__ __launch_bounds__(256) void k3_softpv(
    const float* __restrict__ x, float* __restrict__ out,
    const float* __restrict__ stats, const _Float16* __restrict__ vt,
    const float* __restrict__ beta, const float* __restrict__ gamma)
{
  __shared__ _Float16 P[256 * 64];
  __shared__ float lmx[256], lis[256];

  const int inst = blockIdx.x >> 8;
  const int s    = blockIdx.x & 255;
  const int tid  = threadIdx.x;
  const int lane = tid & 31;
  const int wv   = tid >> 5;
  const int r15  = lane & 15;
  const int hi   = lane >> 4;
  const int kbA  = hi ? 8 : 0;
  const int kbB  = hi ? 16 : 0;

  const float* cost = out + (inst ? C1_OFF : C0_OFF) + (size_t)s * 65536;
  const float* gmx  = stats + (inst ? 131072 : 0) + (size_t)s * 256;
  const float* gsm  = stats + (inst ? 196608 : 65536) + (size_t)s * 256;
  // inst 0 (r2l) weights V_r (side 1); inst 1 (l2r) weights V_l (side 0)
  const _Float16* V = vt + (inst ? 0 : (QH_HALVES / 2)) + (size_t)s * 16384;
  const float* vec  = inst ? gamma : beta;
  const float* xr   = x + ((size_t)inst * 65536 + (size_t)s * 256) * 64;
  float* orow       = out + ((size_t)inst * 65536 + (size_t)s * 256) * 64;

  lmx[tid] = gmx[tid];
  lis[tid] = 1.0f / gsm[tid];

  f8 acc[2][4];
#pragma unroll
  for (int a = 0; a < 2; ++a)
#pragma unroll
    for (int b = 0; b < 4; ++b) acc[a][b] = zero8();

  for (int kc = 0; kc < 4; ++kc) {
    __syncthreads();
    // build normalized P chunk (256 rows x 64 k) as f16 in LDS.
    // Each thread converts one aligned float4 (coalesced b128 reads).
#pragma unroll 4
    for (int it = 0; it < 16; ++it) {
      const int e = it * 1024 + tid * 4;
      const int m = e >> 6, kk = e & 63;
      const f4 v = *(const f4*)(cost + m * 256 + kc * 64 + kk);
      const float mx = lmx[m], is = lis[m];
      h4 t;
#pragma unroll
      for (int q = 0; q < 4; ++q) t[q] = (_Float16)(__expf(v[q] - mx) * is);
      *(h4*)(P + m * 64 + kk) = t;
    }
    __syncthreads();
#pragma unroll
    for (int ks = 0; ks < 2; ++ks) {
      h16 b[4];
#pragma unroll
      for (int nt = 0; nt < 4; ++nt)
        b[nt] = *(const h16*)(V + (size_t)(nt * 16 + r15) * 256 + kc * 64 +
                              ks * 32 + kbB);
#pragma unroll
      for (int tt = 0; tt < 2; ++tt) {
        const _Float16* arow = P + (size_t)(wv * 32 + tt * 16 + r15) * 64 +
                               ks * 32;
        const h16 a = load_a16(arow, kbA);
#pragma unroll
        for (int nt = 0; nt < 4; ++nt) acc[tt][nt] = wmma16(a, b[nt], acc[tt][nt]);
      }
    }
  }

  // epilogue: out = x + F * vec
#pragma unroll
  for (int tt = 0; tt < 2; ++tt) {
#pragma unroll
    for (int nt = 0; nt < 4; ++nt) {
      const int n = nt * 16 + r15;
      const float g = vec[n];
#pragma unroll
      for (int r = 0; r < 8; ++r) {
        const int m = wv * 32 + tt * 16 + hi * 8 + r;
        const int idx = m * 64 + n;
        orow[idx] = xr[idx] + acc[tt][nt][r] * g;
      }
    }
  }
}

// ---------------------------------------------------------------------------
extern "C" void kernel_launch(void* const* d_in, const int* in_sizes, int n_in,
                              void* d_out, int out_size, void* d_ws, size_t ws_size,
                              hipStream_t stream) {
  (void)in_sizes; (void)n_in; (void)out_size; (void)ws_size;
  const float* x    = (const float*)d_in[0];
  const float* c0in = (const float*)d_in[1];
  const float* c1in = (const float*)d_in[2];
  const float* nlw  = (const float*)d_in[3];
  const float* nlb  = (const float*)d_in[4];
  const float* nrw  = (const float*)d_in[5];
  const float* nrb  = (const float*)d_in[6];
  const float* wl1  = (const float*)d_in[7];
  const float* bl1  = (const float*)d_in[8];
  const float* wr1  = (const float*)d_in[9];
  const float* br1  = (const float*)d_in[10];
  const float* wl2  = (const float*)d_in[11];
  const float* bl2  = (const float*)d_in[12];
  const float* wr2  = (const float*)d_in[13];
  const float* br2  = (const float*)d_in[14];
  const float* beta = (const float*)d_in[15];
  const float* gmma = (const float*)d_in[16];

  _Float16* qh = (_Float16*)d_ws;               // 16 MB
  _Float16* vt = qh + QH_HALVES;                // 16 MB
  float* stats = (float*)(vt + QH_HALVES);      // 1 MB
  float* out = (float*)d_out;

  k1_proj<<<512, 256, 0, stream>>>(x, nlw, nlb, nrw, nrb, wl1, bl1, wr1, br1,
                                   wl2, bl2, wr2, br2, qh, vt);
  k2_scores<<<256, 256, 0, stream>>>(qh, c0in, c1in, out, stats);
  k3_softpv<<<512, 256, 0, stream>>>(x, out, stats, vt, beta, gmma);
}
